// CrossDecoderLayer_40793599378097
// MI455X (gfx1250) — matrix operand order repaired and checked by reading, not compiled, not run on hardware
//
#include <hip/hip_runtime.h>
#include <hip/hip_bf16.h>

// ---------------------------------------------------------------------------
// Problem constants (match the reference)
// ---------------------------------------------------------------------------
#define BB 8
#define WW 1024
#define VV 512
#define DD 1024
#define HH 8
#define DKK 128
#define FF 2048
#define SCALE 0.08838834764831845f   // 1/sqrt(128)
#define EPS 1e-5f

typedef __bf16 bf16_t;
typedef __attribute__((ext_vector_type(16))) __bf16 v16bf;
typedef __attribute__((ext_vector_type(8)))  float  v8f;

union AFrag { v16bf v; uint4 q[2]; };
union Frag8 { uint4 q; bf16_t e[8]; };

__device__ __forceinline__ bf16_t f2bf(float f) {
    unsigned u = __builtin_bit_cast(unsigned, f);
    unsigned r = u + 0x7FFFu + ((u >> 16) & 1u);
    unsigned short h = (unsigned short)(r >> 16);
    return __builtin_bit_cast(bf16_t, h);
}

__device__ __forceinline__ v8f vzero8() {
    v8f z;
#pragma unroll
    for (int i = 0; i < 8; ++i) z[i] = 0.0f;
    return z;
}

__device__ __forceinline__ v8f wmma_bf16(v16bf a, v16bf b, v8f c) {
    return __builtin_amdgcn_wmma_f32_16x16x32_bf16(false, a, false, b, (short)0, c,
                                                   false, false);
}

// ---------------------------------------------------------------------------
// f32 -> bf16 cast (vectorized: 4 elements per thread; all sizes are mult of 4)
// ---------------------------------------------------------------------------
__global__ __launch_bounds__(256) void cast_bf16_kernel(const float* __restrict__ in,
                                                        bf16_t* __restrict__ out,
                                                        int n4) {
    int i = blockIdx.x * 256 + threadIdx.x;
    if (i < n4) {
        float4 v = ((const float4*)in)[i];
        union { bf16_t e[4]; uint2 q; } t;
        t.e[0] = f2bf(v.x);
        t.e[1] = f2bf(v.y);
        t.e[2] = f2bf(v.z);
        t.e[3] = f2bf(v.w);
        ((uint2*)out)[i] = t.q;
    }
}

// ---------------------------------------------------------------------------
// LayerNorm over last dim D; one block (256 thr) per row; bf16 output.
// ---------------------------------------------------------------------------
__global__ __launch_bounds__(256) void ln_kernel(const float* __restrict__ x,
                                                 const float* __restrict__ g,
                                                 const float* __restrict__ b,
                                                 bf16_t* __restrict__ outB, int D) {
    const int row = blockIdx.x;
    const float* xr = x + (size_t)row * D;
    __shared__ float red0[8], red1[8], smu, srs;
    float s = 0.f, s2 = 0.f;
    for (int i = threadIdx.x; i < D; i += 256) {
        float v = xr[i];
        s += v;
        s2 += v * v;
    }
#pragma unroll
    for (int m = 16; m >= 1; m >>= 1) {
        s += __shfl_xor(s, m, 32);
        s2 += __shfl_xor(s2, m, 32);
    }
    if ((threadIdx.x & 31) == 0) {
        red0[threadIdx.x >> 5] = s;
        red1[threadIdx.x >> 5] = s2;
    }
    __syncthreads();
    if (threadIdx.x == 0) {
        float ts = 0.f, ts2 = 0.f;
#pragma unroll
        for (int i = 0; i < 8; ++i) {
            ts += red0[i];
            ts2 += red1[i];
        }
        float mu = ts / D;
        float var = ts2 / D - mu * mu;
        smu = mu;
        srs = rsqrtf(var + EPS);
    }
    __syncthreads();
    const float mu = smu, rs = srs;
    for (int i = threadIdx.x; i < D; i += 256)
        outB[(size_t)row * D + i] = f2bf((xr[i] - mu) * rs * g[i] + b[i]);
}

// out = LN(a;g1,b1) + LN(c;g2,b2), f32 out
__global__ __launch_bounds__(256) void dual_ln_add_kernel(
    const float* __restrict__ a, const float* __restrict__ c,
    const float* __restrict__ g1, const float* __restrict__ b1,
    const float* __restrict__ g2, const float* __restrict__ b2,
    float* __restrict__ out, int D) {
    const int row = blockIdx.x;
    const float* ar = a + (size_t)row * D;
    const float* cr = c + (size_t)row * D;
    __shared__ float red[4][8];
    __shared__ float smu0, srs0, smu1, srs1;
    float s0 = 0.f, q0 = 0.f, s1 = 0.f, q1 = 0.f;
    for (int i = threadIdx.x; i < D; i += 256) {
        float va = ar[i], vc = cr[i];
        s0 += va; q0 += va * va;
        s1 += vc; q1 += vc * vc;
    }
#pragma unroll
    for (int m = 16; m >= 1; m >>= 1) {
        s0 += __shfl_xor(s0, m, 32);
        q0 += __shfl_xor(q0, m, 32);
        s1 += __shfl_xor(s1, m, 32);
        q1 += __shfl_xor(q1, m, 32);
    }
    if ((threadIdx.x & 31) == 0) {
        int w = threadIdx.x >> 5;
        red[0][w] = s0; red[1][w] = q0; red[2][w] = s1; red[3][w] = q1;
    }
    __syncthreads();
    if (threadIdx.x == 0) {
        float t0 = 0, u0 = 0, t1 = 0, u1 = 0;
#pragma unroll
        for (int i = 0; i < 8; ++i) {
            t0 += red[0][i]; u0 += red[1][i];
            t1 += red[2][i]; u1 += red[3][i];
        }
        float m0 = t0 / D, m1 = t1 / D;
        smu0 = m0; srs0 = rsqrtf(u0 / D - m0 * m0 + EPS);
        smu1 = m1; srs1 = rsqrtf(u1 / D - m1 * m1 + EPS);
    }
    __syncthreads();
    const float m0 = smu0, r0 = srs0, m1 = smu1, r1 = srs1;
    for (int i = threadIdx.x; i < D; i += 256)
        out[(size_t)row * D + i] =
            (ar[i] - m0) * r0 * g1[i] + b1[i] + (cr[i] - m1) * r1 * g2[i] + b2[i];
}

// ---------------------------------------------------------------------------
// bf16 WMMA GEMM: C[M,N] = A[M,K] @ Bw[K,N] + bias, templated epilogue:
//   MODE 0: out bf16
//   MODE 1: out f32, += resid
//   MODE 2: out bf16, ReLU
// 256 thr = 8 waves; block tile 128x128, K-step 32; wave tile 32x64.
// Register-prefetch double buffering: next global tile loads overlap WMMAs.
// ---------------------------------------------------------------------------
#define TM 128
#define TN 128
#define TK 32

template <int MODE>
__global__ __launch_bounds__(256) void gemm_bf16_kernel(
    const bf16_t* __restrict__ A, const bf16_t* __restrict__ Bw,
    const float* __restrict__ bias, const float* __restrict__ resid,
    float* __restrict__ outF, bf16_t* __restrict__ outB, int M, int N, int K) {
    __shared__ bf16_t As[TM][TK + 8];
    __shared__ bf16_t Bt[TN][TK + 8];   // [n][k]
    const int tid = threadIdx.x;
    const int lane = tid & 31;
    const int wave = tid >> 5;
    const int wrow = wave & 3;   // 0..3 -> 32-row quadrant
    const int wcol = wave >> 2;  // 0..1 -> 64-col half
    const int m0 = blockIdx.y * TM;
    const int n0 = blockIdx.x * TN;
    const int half = lane >> 4;
    const int l15 = lane & 15;

    v8f acc[2][4];
#pragma unroll
    for (int s = 0; s < 2; ++s)
#pragma unroll
        for (int n = 0; n < 4; ++n) acc[s][n] = vzero8();

    const int arow = tid >> 1;
    const int ahalf = tid & 1;
    const int bk = tid >> 3;
    const int bn = (tid & 7) * 16;

    const bf16_t* aptr = A + (size_t)(m0 + arow) * K + ahalf * 16;
    const bf16_t* bptr = Bw + (size_t)bk * N + n0 + bn;

    // prefetch first tile into registers
    uint4 ra0 = *(const uint4*)(aptr);
    uint4 ra1 = *(const uint4*)(aptr + 8);
    uint4 rb0 = *(const uint4*)(bptr);
    uint4 rb1 = *(const uint4*)(bptr + 8);

    for (int k0 = 0; k0 < K; k0 += TK) {
        // stage current tile from registers
        *(uint4*)&As[arow][ahalf * 16] = ra0;
        *(uint4*)&As[arow][ahalf * 16 + 8] = ra1;
        {
            union { uint4 q[2]; bf16_t e[16]; } t;
            t.q[0] = rb0;
            t.q[1] = rb1;
#pragma unroll
            for (int j = 0; j < 16; ++j) Bt[bn + j][bk] = t.e[j];
        }
        __syncthreads();

        // issue next tile's global loads; they overlap the WMMAs below
        if (k0 + TK < K) {
            ra0 = *(const uint4*)(aptr + k0 + TK);
            ra1 = *(const uint4*)(aptr + k0 + TK + 8);
            rb0 = *(const uint4*)(bptr + (size_t)(k0 + TK) * N);
            rb1 = *(const uint4*)(bptr + (size_t)(k0 + TK) * N + 8);
        }

        AFrag af[2];
#pragma unroll
        for (int s = 0; s < 2; ++s) {
            int row = wrow * 32 + s * 16 + l15;
            af[s].q[0] = *(const uint4*)&As[row][half * 8];
            af[s].q[1] = *(const uint4*)&As[row][16 + half * 8];
        }
        AFrag bf[4];
#pragma unroll
        for (int n = 0; n < 4; ++n) {
            int col = wcol * 64 + n * 16 + l15;
            bf[n].q[0] = *(const uint4*)&Bt[col][half * 8];
            bf[n].q[1] = *(const uint4*)&Bt[col][16 + half * 8];
        }
#pragma unroll
        for (int s = 0; s < 2; ++s)
#pragma unroll
            for (int n = 0; n < 4; ++n)
                acc[s][n] = wmma_bf16(af[s].v, bf[n].v, acc[s][n]);
        __syncthreads();
    }

    // branch-free epilogue (MODE is compile-time)
#pragma unroll
    for (int s = 0; s < 2; ++s)
#pragma unroll
        for (int n = 0; n < 4; ++n) {
            const int colg = n0 + wcol * 64 + n * 16 + l15;
            const float bcol = bias[colg];
#pragma unroll
            for (int r = 0; r < 8; ++r) {
                const int rowg = m0 + wrow * 32 + s * 16 + half * 8 + r;
                float v = acc[s][n][r] + bcol;
                if (MODE == 1) {
                    v += resid[(size_t)rowg * N + colg];
                    outF[(size_t)rowg * N + colg] = v;
                } else {
                    if (MODE == 2) v = fmaxf(v, 0.0f);
                    outB[(size_t)rowg * N + colg] = f2bf(v);
                }
            }
        }
}

// ---------------------------------------------------------------------------
// Self-attention (flash-style, masked softmax).
// grid (W/64, H, B), 128 thr = 4 waves; each wave owns 16 q rows.
// Q,K,V: bf16 [B,W,H,DK]. O: bf16 [B,W,H,DK].
// ---------------------------------------------------------------------------
__global__ __launch_bounds__(128) void self_attn_kernel(
    const bf16_t* __restrict__ Q, const bf16_t* __restrict__ Kb,
    const bf16_t* __restrict__ Vb, const unsigned char* __restrict__ mask,
    bf16_t* __restrict__ O) {
    const int qt = blockIdx.x, h = blockIdx.y, b = blockIdx.z;
    const int tid = threadIdx.x, lane = tid & 31, wave = tid >> 5;
    const int half = lane >> 4, l15 = lane & 15;

    __shared__ bf16_t Ks[64][DKK + 16];       // [key][dk]
    __shared__ bf16_t Vt[DKK][64 + 16];       // [dk][key]  (transposed)
    __shared__ bf16_t Ps[4][16][64 + 16];     // per-wave probs

    const int qrow0 = qt * 64 + wave * 16;
    const unsigned char* mbase = mask + (size_t)b * WW * WW;

    AFrag qf[4];
    {
        int row = qrow0 + l15;
        const bf16_t* qp = Q + (((size_t)(b * WW + row)) * HH + h) * DKK;
#pragma unroll
        for (int c = 0; c < 4; ++c) {
            qf[c].q[0] = *(const uint4*)(qp + c * 32 + half * 8);
            qf[c].q[1] = *(const uint4*)(qp + c * 32 + 16 + half * 8);
        }
    }

    float mrow[8], lrow[8];
    v8f Oa[8];
#pragma unroll
    for (int r = 0; r < 8; ++r) { mrow[r] = -1e30f; lrow[r] = 0.0f; }
#pragma unroll
    for (int n = 0; n < 8; ++n) Oa[n] = vzero8();

    const int srow = tid >> 1;           // staging row 0..63
    const int soff = (tid & 1) * 64;     // staging dk half

    for (int kt = 0; kt < WW / 64; ++kt) {
        __syncthreads();
        {   // stage K tile
            const bf16_t* src =
                Kb + (((size_t)(b * WW + kt * 64 + srow)) * HH + h) * DKK + soff;
#pragma unroll
            for (int j = 0; j < 8; ++j)
                ((uint4*)&Ks[srow][soff])[j] = ((const uint4*)src)[j];
        }
        {   // stage V tile transposed
            const bf16_t* src =
                Vb + (((size_t)(b * WW + kt * 64 + srow)) * HH + h) * DKK + soff;
#pragma unroll
            for (int d0 = 0; d0 < 64; d0 += 8) {
                Frag8 t;
                t.q = *(const uint4*)(src + d0);
#pragma unroll
                for (int j = 0; j < 8; ++j) Vt[soff + d0 + j][srow] = t.e[j];
            }
        }
        __syncthreads();

        // S = Q K^T  (16 x 64)
        v8f S[4];
#pragma unroll
        for (int n = 0; n < 4; ++n) S[n] = vzero8();
#pragma unroll
        for (int n = 0; n < 4; ++n) {
            int col = n * 16 + l15;  // key in tile
#pragma unroll
            for (int c = 0; c < 4; ++c) {
                AFrag kf;
                kf.q[0] = *(const uint4*)&Ks[col][c * 32 + half * 8];
                kf.q[1] = *(const uint4*)&Ks[col][c * 32 + 16 + half * 8];
                S[n] = wmma_bf16(qf[c].v, kf.v, S[n]);
            }
        }
        // scale + mask
#pragma unroll
        for (int n = 0; n < 4; ++n)
#pragma unroll
            for (int r = 0; r < 8; ++r) {
                int qr = qrow0 + half * 8 + r;
                int kc = kt * 64 + n * 16 + l15;
                float sv = S[n][r] * SCALE;
                if (!mbase[(size_t)qr * WW + kc]) sv = -1e9f;
                S[n][r] = sv;
            }
        // online softmax
        float mnew[8], alpha[8];
#pragma unroll
        for (int r = 0; r < 8; ++r) {
            float mx = S[0][r];
#pragma unroll
            for (int n = 1; n < 4; ++n) mx = fmaxf(mx, S[n][r]);
#pragma unroll
            for (int d = 1; d < 16; d <<= 1) mx = fmaxf(mx, __shfl_xor(mx, d, 32));
            mnew[r] = fmaxf(mx, mrow[r]);
            alpha[r] = __expf(mrow[r] - mnew[r]);
        }
#pragma unroll
        for (int n = 0; n < 4; ++n)
#pragma unroll
            for (int r = 0; r < 8; ++r) S[n][r] = __expf(S[n][r] - mnew[r]);
#pragma unroll
        for (int r = 0; r < 8; ++r) {
            float sm = 0.f;
#pragma unroll
            for (int n = 0; n < 4; ++n) sm += S[n][r];
#pragma unroll
            for (int d = 1; d < 16; d <<= 1) sm += __shfl_xor(sm, d, 32);
            lrow[r] = lrow[r] * alpha[r] + sm;
            mrow[r] = mnew[r];
        }
#pragma unroll
        for (int n = 0; n < 8; ++n)
#pragma unroll
            for (int r = 0; r < 8; ++r) Oa[n][r] *= alpha[r];
        // stash P as bf16 for A-fragment layout
#pragma unroll
        for (int n = 0; n < 4; ++n)
#pragma unroll
            for (int r = 0; r < 8; ++r)
                Ps[wave][half * 8 + r][n * 16 + l15] = f2bf(S[n][r]);
        __syncthreads();
        // O += P V
        AFrag pf[2];
#pragma unroll
        for (int c = 0; c < 2; ++c) {
            pf[c].q[0] = *(const uint4*)&Ps[wave][l15][c * 32 + half * 8];
            pf[c].q[1] = *(const uint4*)&Ps[wave][l15][c * 32 + 16 + half * 8];
        }
#pragma unroll
        for (int nt = 0; nt < 8; ++nt) {
            int col = nt * 16 + l15;  // dk
#pragma unroll
            for (int c = 0; c < 2; ++c) {
                AFrag vf;
                vf.q[0] = *(const uint4*)&Vt[col][c * 32 + half * 8];
                vf.q[1] = *(const uint4*)&Vt[col][c * 32 + 16 + half * 8];
                Oa[nt] = wmma_bf16(pf[c].v, vf.v, Oa[nt]);
            }
        }
    }
    // normalize + write out
#pragma unroll
    for (int nt = 0; nt < 8; ++nt)
#pragma unroll
        for (int r = 0; r < 8; ++r) {
            int qr = qrow0 + half * 8 + r;
            int dk = nt * 16 + l15;
            float v = Oa[nt][r] / lrow[r];
            O[(((size_t)(b * WW + qr)) * HH + h) * DKK + dk] = f2bf(v);
        }
}

// ---------------------------------------------------------------------------
// Sigmoid-gated cross attention.
// grid (W/16, B), 128 thr = 4 waves; wave owns v-tiles {wave, wave+4} (64 keys
// each), loops over all 8 heads. Partial res accumulated via global f32
// atomics (res pre-zeroed); gate head-sum kept in registers -> attn_out.
// ---------------------------------------------------------------------------
__global__ __launch_bounds__(128) void cross_attn_kernel(
    const bf16_t* __restrict__ CQ, const bf16_t* __restrict__ CK,
    const bf16_t* __restrict__ CV, float* __restrict__ res,
    float* __restrict__ attn_out) {
    extern __shared__ char smem[];
    const int wt = blockIdx.x, b = blockIdx.y;
    const int tid = threadIdx.x, lane = tid & 31, wave = tid >> 5;
    const int half = lane >> 4, l15 = lane & 15;

    // per-wave scratch
    bf16_t(*cvT)[80] = (bf16_t(*)[80])(smem + (size_t)wave * (DKK * 80 * 2));
    bf16_t(*Ps)[80] =
        (bf16_t(*)[80])(smem + 4 * (size_t)(DKK * 80 * 2) + (size_t)wave * (16 * 80 * 2));

    const int w0 = wt * 16;
    v8f gsum[2][4];
#pragma unroll
    for (int vi = 0; vi < 2; ++vi)
#pragma unroll
        for (int n = 0; n < 4; ++n) gsum[vi][n] = vzero8();

    for (int h = 0; h < HH; ++h) {
        AFrag qf[4];
        {
            int row = w0 + l15;
            const bf16_t* qp = CQ + (((size_t)(b * WW + row)) * HH + h) * DKK;
#pragma unroll
            for (int c = 0; c < 4; ++c) {
                qf[c].q[0] = *(const uint4*)(qp + c * 32 + half * 8);
                qf[c].q[1] = *(const uint4*)(qp + c * 32 + 16 + half * 8);
            }
        }
        v8f Oa[8];
#pragma unroll
        for (int n = 0; n < 8; ++n) Oa[n] = vzero8();

        for (int vi = 0; vi < 2; ++vi) {
            const int vt = wave + vi * 4;  // key tile (64 keys)
            // stage cv^T (wave-local): [dk][key]
#pragma unroll
            for (int kk = 0; kk < 2; ++kk) {
                int key = lane * 2 + kk;
                const bf16_t* src =
                    CV + (((size_t)(b * VV + vt * 64 + key)) * HH + h) * DKK;
#pragma unroll
                for (int d0 = 0; d0 < DKK; d0 += 8) {
                    Frag8 t;
                    t.q = *(const uint4*)(src + d0);
#pragma unroll
                    for (int j = 0; j < 8; ++j) cvT[d0 + j][key] = t.e[j];
                }
            }
            __syncthreads();
            // S = cq ck^T (16 x 64), B-frags straight from global (contiguous dk)
            v8f S[4];
#pragma unroll
            for (int n = 0; n < 4; ++n) S[n] = vzero8();
#pragma unroll
            for (int n = 0; n < 4; ++n) {
                int keyrow = vt * 64 + n * 16 + l15;
                const bf16_t* kp =
                    CK + (((size_t)(b * VV + keyrow)) * HH + h) * DKK;
#pragma unroll
                for (int c = 0; c < 4; ++c) {
                    AFrag kf;
                    kf.q[0] = *(const uint4*)(kp + c * 32 + half * 8);
                    kf.q[1] = *(const uint4*)(kp + c * 32 + 16 + half * 8);
                    S[n] = wmma_bf16(qf[c].v, kf.v, S[n]);
                }
            }
            // sigmoid gate; accumulate head-sum; stash bf16 P
#pragma unroll
            for (int n = 0; n < 4; ++n)
#pragma unroll
                for (int r = 0; r < 8; ++r) {
                    float g = 1.0f / (1.0f + __expf(-S[n][r] * SCALE));
                    S[n][r] = g;
                    gsum[vi][n][r] += g;
                    Ps[half * 8 + r][n * 16 + l15] = f2bf(g);
                }
            __syncthreads();
            // Oa += gate * cv
            AFrag pf[2];
#pragma unroll
            for (int c = 0; c < 2; ++c) {
                pf[c].q[0] = *(const uint4*)&Ps[l15][c * 32 + half * 8];
                pf[c].q[1] = *(const uint4*)&Ps[l15][c * 32 + 16 + half * 8];
            }
#pragma unroll
            for (int nt = 0; nt < 8; ++nt) {
                int col = nt * 16 + l15;  // dk
#pragma unroll
                for (int c = 0; c < 2; ++c) {
                    AFrag vf;
                    vf.q[0] = *(const uint4*)&cvT[col][c * 32 + half * 8];
                    vf.q[1] = *(const uint4*)&cvT[col][c * 32 + 16 + half * 8];
                    Oa[nt] = wmma_bf16(pf[c].v, vf.v, Oa[nt]);
                }
            }
            __syncthreads();
        }
        // accumulate partial res for this head
#pragma unroll
        for (int nt = 0; nt < 8; ++nt)
#pragma unroll
            for (int r = 0; r < 8; ++r) {
                int wr = w0 + half * 8 + r;
                int dk = nt * 16 + l15;
                atomicAdd(&res[((size_t)(b * WW + wr)) * DD + h * DKK + dk],
                          Oa[nt][r]);
            }
    }
    // attn_out = mean over heads of gate
#pragma unroll
    for (int vi = 0; vi < 2; ++vi)
#pragma unroll
        for (int n = 0; n < 4; ++n)
#pragma unroll
            for (int r = 0; r < 8; ++r) {
                int wr = w0 + half * 8 + r;
                int vc = (wave + vi * 4) * 64 + n * 16 + l15;
                attn_out[((size_t)(b * WW + wr)) * VV + vc] =
                    gsum[vi][n][r] * (1.0f / HH);
            }
}

// ---------------------------------------------------------------------------
// Orchestration
// ---------------------------------------------------------------------------
extern "C" void kernel_launch(void* const* d_in, const int* in_sizes, int n_in,
                              void* d_out, int out_size, void* d_ws, size_t ws_size,
                              hipStream_t stream) {
    (void)in_sizes; (void)n_in; (void)out_size; (void)ws_size;
    const size_t MW = (size_t)BB * WW;   // 8192
    const size_t MV = (size_t)BB * VV;   // 4096

    const float* x = (const float*)d_in[0];
    const float* y = (const float*)d_in[1];
    const unsigned char* mask = (const unsigned char*)d_in[2];

    char* ws = (char*)d_ws;
    size_t off = 0;
    auto take = [&](size_t bytes) -> char* {
        char* p = ws + off;
        off += (bytes + 255) & ~(size_t)255;
        return p;
    };
    // bf16 weights
    bf16_t* wq  = (bf16_t*)take((size_t)DD * DD * 2);
    bf16_t* wk  = (bf16_t*)take((size_t)DD * DD * 2);
    bf16_t* wv  = (bf16_t*)take((size_t)DD * DD * 2);
    bf16_t* wo  = (bf16_t*)take((size_t)DD * DD * 2);
    bf16_t* cwq = (bf16_t*)take((size_t)DD * DD * 2);
    bf16_t* cwk = (bf16_t*)take((size_t)DD * DD * 2);
    bf16_t* cwv = (bf16_t*)take((size_t)DD * DD * 2);
    bf16_t* w1  = (bf16_t*)take((size_t)DD * FF * 2);
    bf16_t* w2  = (bf16_t*)take((size_t)FF * DD * 2);
    // activations (with reuse)
    bf16_t* hbf = (bf16_t*)take(MW * DD * 2);   // h / t / h2
    bf16_t* qbf = (bf16_t*)take(MW * DD * 2);   // q / cq
    bf16_t* kbf = (bf16_t*)take(MW * DD * 2);   // k / (ck|cv)
    bf16_t* vbf = (bf16_t*)take(MW * DD * 2);   // v / u-low (contiguous with obf)
    bf16_t* obf = (bf16_t*)take(MW * DD * 2);   // o / u-high
    float*  x1  = (float*)take(MW * DD * 4);
    float*  rsb = (float*)take(MW * DD * 4);
    float*  x2  = (float*)take(MW * DD * 4);
    bf16_t* avb = (bf16_t*)take(MV * DD * 2);
    bf16_t* ckb = kbf;
    bf16_t* cvb = kbf + MV * DD;
    bf16_t* ubf = vbf;                          // spans vbf+obf = MW*FF bf16

    float* outx = (float*)d_out;
    float* outg = (float*)d_out + MW * DD;

    auto cast = [&](int idx, bf16_t* dst, size_t n) {
        int n4 = (int)(n / 4);
        cast_bf16_kernel<<<dim3((unsigned)((n4 + 255) / 256)), 256, 0, stream>>>(
            (const float*)d_in[idx], dst, n4);
    };
    cast(5, wq, (size_t)DD * DD);
    cast(7, wk, (size_t)DD * DD);
    cast(9, wv, (size_t)DD * DD);
    cast(11, wo, (size_t)DD * DD);
    cast(17, cwq, (size_t)DD * DD);
    cast(19, cwk, (size_t)DD * DD);
    cast(21, cwv, (size_t)DD * DD);
    cast(29, w1, (size_t)DD * FF);
    cast(31, w2, (size_t)FF * DD);

    // --- res1: h = LN(x); q,k,v; self-attn; x1 = x + o@wo + bo
    ln_kernel<<<dim3((unsigned)MW), 256, 0, stream>>>(
        x, (const float*)d_in[3], (const float*)d_in[4], hbf, DD);
    gemm_bf16_kernel<0><<<dim3(DD / TN, (unsigned)MW / TM), 256, 0, stream>>>(
        hbf, wq, (const float*)d_in[6], nullptr, nullptr, qbf, (int)MW, DD, DD);
    gemm_bf16_kernel<0><<<dim3(DD / TN, (unsigned)MW / TM), 256, 0, stream>>>(
        hbf, wk, (const float*)d_in[8], nullptr, nullptr, kbf, (int)MW, DD, DD);
    gemm_bf16_kernel<0><<<dim3(DD / TN, (unsigned)MW / TM), 256, 0, stream>>>(
        hbf, wv, (const float*)d_in[10], nullptr, nullptr, vbf, (int)MW, DD, DD);
    self_attn_kernel<<<dim3(WW / 64, HH, BB), 128, 0, stream>>>(qbf, kbf, vbf,
                                                                mask, obf);
    gemm_bf16_kernel<1><<<dim3(DD / TN, (unsigned)MW / TM), 256, 0, stream>>>(
        obf, wo, (const float*)d_in[12], x, x1, nullptr, (int)MW, DD, DD);

    // --- cross attention
    ln_kernel<<<dim3((unsigned)MW), 256, 0, stream>>>(
        x1, (const float*)d_in[13], (const float*)d_in[14], hbf, DD);
    ln_kernel<<<dim3((unsigned)MV), 256, 0, stream>>>(
        y, (const float*)d_in[15], (const float*)d_in[16], avb, DD);
    gemm_bf16_kernel<0><<<dim3(DD / TN, (unsigned)MW / TM), 256, 0, stream>>>(
        hbf, cwq, (const float*)d_in[18], nullptr, nullptr, qbf, (int)MW, DD, DD);
    gemm_bf16_kernel<0><<<dim3(DD / TN, (unsigned)MV / TM), 256, 0, stream>>>(
        avb, cwk, (const float*)d_in[20], nullptr, nullptr, ckb, (int)MV, DD, DD);
    gemm_bf16_kernel<0><<<dim3(DD / TN, (unsigned)MV / TM), 256, 0, stream>>>(
        avb, cwv, (const float*)d_in[22], nullptr, nullptr, cvb, (int)MV, DD, DD);
    (void)hipMemsetAsync(rsb, 0, MW * DD * 4, stream);
    {
        size_t dynlds = 4 * (size_t)(DKK * 80 * 2) + 4 * (size_t)(16 * 80 * 2);
        cross_attn_kernel<<<dim3(WW / 16, BB), 128, dynlds, stream>>>(
            qbf, ckb, cvb, rsb, outg);
    }

    // --- x2 = LN(x1;n1) + LN(res;n2)
    dual_ln_add_kernel<<<dim3((unsigned)MW), 256, 0, stream>>>(
        x1, rsb, (const float*)d_in[23], (const float*)d_in[24],
        (const float*)d_in[25], (const float*)d_in[26], x2, DD);

    // --- res2: FFN
    ln_kernel<<<dim3((unsigned)MW), 256, 0, stream>>>(
        x2, (const float*)d_in[27], (const float*)d_in[28], hbf, DD);
    gemm_bf16_kernel<2><<<dim3(FF / TN, (unsigned)MW / TM), 256, 0, stream>>>(
        hbf, w1, (const float*)d_in[30], nullptr, nullptr, ubf, (int)MW, FF, DD);
    gemm_bf16_kernel<1><<<dim3(DD / TN, (unsigned)MW / TM), 256, 0, stream>>>(
        ubf, w2, (const float*)d_in[32], x2, outx, nullptr, (int)MW, DD, FF);
}
